// MultiHeadSelfAttention_15109694947882
// MI455X (gfx1250) — matrix operand order repaired and checked by reading, not compiled, not run on hardware
//
#include <hip/hip_runtime.h>

// ---------------------------------------------------------------------------
// MultiHeadSelfAttention for MI455X (gfx1250, wave32, WMMA bf16 -> f32 acc)
// L=4096, N=1, E=1024, H=16, DH=64
// Async (GLOBAL_LOAD_ASYNC_TO_LDS_B128) double-buffered staging + WMMA.
// ---------------------------------------------------------------------------

#define L_DIM 4096
#define E_DIM 1024
#define H_DIM 16
#define DH_DIM 64
#define QK_SCALE 0.125f  // 64^-0.5

typedef __attribute__((ext_vector_type(16))) __bf16 v16bf;
typedef __attribute__((ext_vector_type(8)))  float  v8f;

union FragU { v16bf v; uint4 q[2]; };

__device__ __forceinline__ unsigned short f2bf(float f) {
  unsigned int u = __float_as_uint(f);
  u += 0x7FFFu + ((u >> 16) & 1u);   // round-to-nearest-even
  return (unsigned short)(u >> 16);
}

// ---- CDNA5 async copy: global -> LDS, 16 bytes per lane, ASYNCcnt-tracked --
__device__ __forceinline__ unsigned lds_addr(const void* p) {
  // generic LDS pointers carry the 32-bit LDS byte offset in the low word
  return (unsigned)(size_t)p;
}
__device__ __forceinline__ void async_b128(unsigned ldsA, const void* g) {
  unsigned long long ga = (unsigned long long)(size_t)g;
  asm volatile("global_load_async_to_lds_b128 %0, %1, off"
               :: "v"(ldsA), "v"(ga) : "memory");
}
__device__ __forceinline__ void wait_async0() {
  asm volatile("s_wait_asynccnt 0x0" ::: "memory");
}

// A-matrix fragment (16x32 bf16, M across lanes 0..15, K split per lane-half):
// lane<16 holds K {0..7, 16..23}; lane>=16 holds K {8..15, 24..31}
__device__ __forceinline__ v16bf frag_a(const unsigned short* base, int rowShorts, int kOff) {
  int lane = threadIdx.x & 31;
  const unsigned short* p = base + (lane & 15) * rowShorts + kOff + ((lane >> 4) << 3);
  FragU f;
  f.q[0] = *(const uint4*)(p);
  f.q[1] = *(const uint4*)(p + 16);
  return f.v;
}

// B-matrix fragment (32x16 bf16), LDS stored N-major (row = n, rowShorts per row):
// lane<16 holds K {0..7, 16..23}; lane>=16 holds K {8..15, 24..31}
__device__ __forceinline__ v16bf frag_b(const unsigned short* base, int rowShorts, int kOff) {
  int lane = threadIdx.x & 31;
  const unsigned short* p = base + (lane & 15) * rowShorts + kOff + ((lane >> 4) << 3);
  FragU f;
  f.q[0] = *(const uint4*)(p);
  f.q[1] = *(const uint4*)(p + 16);
  return f.v;
}

__device__ __forceinline__ v8f wmma_bf16(v16bf a, v16bf b, v8f c) {
  return __builtin_amdgcn_wmma_f32_16x16x32_bf16(false, a, false, b, (short)0, c,
                                                 false, false);
}

// ---------------------------------------------------------------------------
// f32 -> bf16 cast, 4-wide
// ---------------------------------------------------------------------------
__global__ __launch_bounds__(256) void cast_f32_bf16(const float* __restrict__ in,
                                                     unsigned short* __restrict__ out,
                                                     int n4) {
  int i = blockIdx.x * 256 + threadIdx.x;
  if (i >= n4) return;
  float4 f = ((const float4*)in)[i];
  ushort4 o;
  o.x = f2bf(f.x); o.y = f2bf(f.y); o.z = f2bf(f.z); o.w = f2bf(f.w);
  ((ushort4*)out)[i] = o;
}

// ---------------------------------------------------------------------------
// bf16 GEMM: C[m][n] = sum_k A[m][k] * B[n][k]   (both row-major over k)
// 128x128 tile, K-step 32, 256 threads = 8 waves; double-buffered async LDS.
// mode 0: write f32 to outF[m*N + n]
// mode 1: QKV scatter -> Q[H][L][DH]*scale, K[H][L][DH], Vt[H][DH][L] (bf16)
// ---------------------------------------------------------------------------
__global__ __launch_bounds__(256) void gemm_bf16(const unsigned short* __restrict__ A,
                                                 const unsigned short* __restrict__ B,
                                                 int M, int N, int K, int nBlocksN,
                                                 int mode, float scale,
                                                 float* __restrict__ outF,
                                                 unsigned short* __restrict__ Qb,
                                                 unsigned short* __restrict__ Kb,
                                                 unsigned short* __restrict__ Vtb) {
  __shared__ unsigned short As[2][128 * 32];
  __shared__ unsigned short Bs[2][128 * 32];

  const int bid  = blockIdx.x;
  const int bm   = bid / nBlocksN;
  const int bn   = bid % nBlocksN;
  const int tid  = threadIdx.x;
  const int wave = tid >> 5;
  const int lane = tid & 31;
  const int half = lane >> 4;
  const int col  = lane & 15;

  const unsigned ldsA = lds_addr(&As[0][0]);
  const unsigned ldsB = lds_addr(&Bs[0][0]);

  v8f acc[8];
  const v8f vzero = {0.f, 0.f, 0.f, 0.f, 0.f, 0.f, 0.f, 0.f};
#pragma unroll
  for (int i = 0; i < 8; i++) acc[i] = vzero;

  const unsigned short* Ab = A + (size_t)(bm * 128) * K;
  const unsigned short* Bb = B + (size_t)(bn * 128) * K;

  // stage one 128x32 bf16 tile pair into LDS buffer `buf` (async, 4x b128/thread)
  auto stageAB = [&](int kt, int buf) {
    int kc = kt * 32;
    unsigned aL = ldsA + (unsigned)buf * 8192u;
    unsigned bL = ldsB + (unsigned)buf * 8192u;
#pragma unroll
    for (int u = 0; u < 2; u++) {
      int c = tid + u * 256;             // chunk id, 512 chunks of 8 shorts
      int row = c >> 2, cofs = (c & 3) << 3;
      async_b128(aL + ((unsigned)c << 4), Ab + (size_t)row * K + kc + cofs);
      async_b128(bL + ((unsigned)c << 4), Bb + (size_t)row * K + kc + cofs);
    }
  };

  const int nKT = K >> 5;
  stageAB(0, 0);
  wait_async0();
  __syncthreads();

  for (int kt = 0; kt < nKT; kt++) {
    int buf = kt & 1;
    if (kt + 1 < nKT) stageAB(kt + 1, buf ^ 1);   // overlap DMA with WMMAs

    v16bf af = frag_a(&As[buf][0] + wave * 16 * 32, 32, 0);
#pragma unroll
    for (int nt = 0; nt < 8; nt++) {
      v16bf bf_ = frag_b(&Bs[buf][0] + nt * 16 * 32, 32, 0);
      acc[nt] = wmma_bf16(af, bf_, acc[nt]);
    }

    wait_async0();      // own async tile (kt+1) landed
    __syncthreads();    // everyone done reading buf, everyone's DMA visible
  }

  // epilogue: C layout — VGPR r holds (m = r + half*8, n = col) of each 16x16 tile
  const int m0 = bm * 128 + wave * 16 + half * 8;
#pragma unroll
  for (int nt = 0; nt < 8; nt++) {
    int n = bn * 128 + nt * 16 + col;
#pragma unroll
    for (int r = 0; r < 8; r++) {
      float v = acc[nt][r];
      int m = m0 + r;
      if (mode == 0) {
        outF[(size_t)m * N + n] = v;
      } else {
        int which = n >> 10, f = n & 1023, h = f >> 6, d = f & 63;
        if (which == 0)
          Qb[((size_t)(h * L_DIM + m) << 6) + d] = f2bf(v * scale);
        else if (which == 1)
          Kb[((size_t)(h * L_DIM + m) << 6) + d] = f2bf(v);
        else
          Vtb[((size_t)(h * DH_DIM + d) << 12) + m] = f2bf(v);
      }
    }
  }
}

// ---------------------------------------------------------------------------
// Flash attention: one block = (head h, 64 query rows); 4 waves x 16 rows.
// Async double-buffered 32-key K/V tiles; online softmax; WMMA QK^T and PV.
// Writes O (softmax(QK^T)V) as bf16 to Ob[L][E] with column = h*64 + d.
// ---------------------------------------------------------------------------
__global__ __launch_bounds__(128) void attn_kernel(const unsigned short* __restrict__ Qb,
                                                   const unsigned short* __restrict__ Kb,
                                                   const unsigned short* __restrict__ Vtb,
                                                   unsigned short* __restrict__ Ob) {
  __shared__ unsigned short Qs[64 * 64];      // [qrow][d]
  __shared__ unsigned short Ks[2][32 * 64];   // [key][d]  (N-major for S B-operand)
  __shared__ unsigned short Vs[2][64 * 32];   // [d][key]  (N-major for PV B-operand)
  __shared__ unsigned short Ps[4][16 * 32];   // per-wave P re-layout scratch

  const int h    = blockIdx.x & 15;
  const int qblk = blockIdx.x >> 4;           // 0..63
  const int tid  = threadIdx.x;
  const int wave = tid >> 5;
  const int lane = tid & 31;
  const int half = lane >> 4;
  const int col  = lane & 15;

  const unsigned ldsQ = lds_addr(&Qs[0]);
  const unsigned ldsK = lds_addr(&Ks[0][0]);
  const unsigned ldsV = lds_addr(&Vs[0][0]);

  const unsigned short* Qg = Qb + (((size_t)h * L_DIM + qblk * 64) << 6);
  const unsigned short* Kg0 = Kb + (((size_t)h * L_DIM) << 6);
  const unsigned short* Vg0 = Vtb + (((size_t)h * DH_DIM) << 12);

  // stage K[32x64] (contiguous) and Vt[64x32] (row stride L) tile j -> buf
  auto stageKV = [&](int j, int buf) {
    const unsigned short* Kg = Kg0 + ((size_t)(j * 32) << 6);
    const unsigned short* Vg = Vg0 + j * 32;
    unsigned kL = ldsK + (unsigned)buf * 4096u;
    unsigned vL = ldsV + (unsigned)buf * 4096u;
#pragma unroll
    for (int u = 0; u < 2; u++) {
      int c = tid + u * 128;               // 256 chunks of 8 shorts each
      async_b128(kL + ((unsigned)c << 4), Kg + c * 8);
      int d = c >> 2, cofs = (c & 3) << 3;
      async_b128(vL + ((unsigned)c << 4), Vg + ((size_t)d << 12) + cofs);
    }
  };

  // prologue: Q tile (4x b128/thread) + first K/V tile, all async
#pragma unroll
  for (int u = 0; u < 4; u++) {
    int c = tid + u * 128;                 // 512 chunks
    async_b128(ldsQ + ((unsigned)c << 4), Qg + c * 8);
  }
  stageKV(0, 0);
  wait_async0();
  __syncthreads();

  const v16bf qf0 = frag_a(Qs + wave * 16 * 64, 64, 0);
  const v16bf qf1 = frag_a(Qs + wave * 16 * 64, 64, 32);

  const v8f vzero = {0.f, 0.f, 0.f, 0.f, 0.f, 0.f, 0.f, 0.f};
  v8f o[4];
#pragma unroll
  for (int i = 0; i < 4; i++) o[i] = vzero;
  float mrow[8], lrow[8];
#pragma unroll
  for (int r = 0; r < 8; r++) { mrow[r] = -1.0e30f; lrow[r] = 0.f; }

  const int nJ = L_DIM / 32;
  for (int j = 0; j < nJ; j++) {
    int buf = j & 1;
    if (j + 1 < nJ) stageKV(j + 1, buf ^ 1);   // DMA next tile under compute

    // S = Q K^T : two 16x16 key tiles, K-dim = DH = 64 (two 32-chunks)
    v8f s[2];
#pragma unroll
    for (int t = 0; t < 2; t++) {
      v16bf kf0 = frag_b(&Ks[buf][0] + t * 16 * 64, 64, 0);
      v16bf kf1 = frag_b(&Ks[buf][0] + t * 16 * 64, 64, 32);
      s[t] = wmma_bf16(qf0, kf0, vzero);
      s[t] = wmma_bf16(qf1, kf1, s[t]);
    }

    // online softmax: VGPR r <-> row (r + half*8); butterfly over 16 lanes
    float mnew[8];
#pragma unroll
    for (int r = 0; r < 8; r++) {
      float v = fmaxf(s[0][r], s[1][r]);
#pragma unroll
      for (int mk = 1; mk < 16; mk <<= 1) v = fmaxf(v, __shfl_xor(v, mk, 32));
      mnew[r] = fmaxf(mrow[r], v);
    }
#pragma unroll
    for (int r = 0; r < 8; r++) {
      float corr = __expf(mrow[r] - mnew[r]);
      mrow[r] = mnew[r];
      float p0 = __expf(s[0][r] - mnew[r]);
      float p1 = __expf(s[1][r] - mnew[r]);
      float rs = p0 + p1;
#pragma unroll
      for (int mk = 1; mk < 16; mk <<= 1) rs += __shfl_xor(rs, mk, 32);
      lrow[r] = lrow[r] * corr + rs;
      o[0][r] *= corr; o[1][r] *= corr; o[2][r] *= corr; o[3][r] *= corr;
      int prow = r + half * 8;
      Ps[wave][prow * 32 + col]      = f2bf(p0);
      Ps[wave][prow * 32 + col + 16] = f2bf(p1);
    }
    asm volatile("s_wait_dscnt 0x0" ::: "memory");  // P stores -> P fragment loads

    // O += P V : A = P (16 x 32 keys), B = V tile (32 keys x 64 d), 4 d-tiles
    v16bf pf = frag_a(Ps[wave], 32, 0);
#pragma unroll
    for (int nt = 0; nt < 4; nt++) {
      v16bf vf = frag_b(&Vs[buf][0] + nt * 16 * 32, 32, 0);
      o[nt] = wmma_bf16(pf, vf, o[nt]);
    }

    wait_async0();      // own async tile (j+1) landed in LDS
    __syncthreads();    // all reads of buf done, all DMAs visible
  }

  // normalize and store bf16 into Ob[L][E] at column h*64 + d
#pragma unroll
  for (int r = 0; r < 8; r++) {
    float inv = 1.0f / lrow[r];
    int m = qblk * 64 + wave * 16 + half * 8 + r;
#pragma unroll
    for (int nt = 0; nt < 4; nt++) {
      Ob[(size_t)m * E_DIM + h * 64 + nt * 16 + col] = f2bf(o[nt][r] * inv);
    }
  }
}

// ---------------------------------------------------------------------------
// launch
// ---------------------------------------------------------------------------
extern "C" void kernel_launch(void* const* d_in, const int* in_sizes, int n_in,
                              void* d_out, int out_size, void* d_ws, size_t ws_size,
                              hipStream_t stream) {
  const float* x     = (const float*)d_in[0];  // [L, 1, E]
  const float* wqkv  = (const float*)d_in[1];  // [3E, E]
  const float* wout  = (const float*)d_in[2];  // [E, E]
  float* out = (float*)d_out;                  // [L, 1, E]

  unsigned short* ws = (unsigned short*)d_ws;
  size_t off = 0;
  unsigned short* xb  = ws + off; off += (size_t)L_DIM * E_DIM;       // x bf16
  unsigned short* wqb = ws + off; off += (size_t)3 * E_DIM * E_DIM;   // w_qkv bf16
  unsigned short* wob = ws + off; off += (size_t)E_DIM * E_DIM;       // w_out bf16
  unsigned short* Qb  = ws + off; off += (size_t)L_DIM * E_DIM;       // Q[H][L][DH]
  unsigned short* Kb  = ws + off; off += (size_t)L_DIM * E_DIM;       // K[H][L][DH]
  unsigned short* Vtb = ws + off; off += (size_t)L_DIM * E_DIM;       // V^T[H][DH][L]
  unsigned short* Ob  = ws + off; off += (size_t)L_DIM * E_DIM;       // attn out bf16
  // total: 24M shorts = 48 MB of workspace

  cast_f32_bf16<<<(L_DIM * E_DIM / 4 + 255) / 256, 256, 0, stream>>>(x, xb, L_DIM * E_DIM / 4);
  cast_f32_bf16<<<(3 * E_DIM * E_DIM / 4 + 255) / 256, 256, 0, stream>>>(wqkv, wqb, 3 * E_DIM * E_DIM / 4);
  cast_f32_bf16<<<(E_DIM * E_DIM / 4 + 255) / 256, 256, 0, stream>>>(wout, wob, E_DIM * E_DIM / 4);

  // QKV projection: M=4096, N=3072, K=1024 -> 32 x 24 tiles
  gemm_bf16<<<32 * 24, 256, 0, stream>>>(xb, wqb, L_DIM, 3 * E_DIM, E_DIM, 24,
                                         /*mode=*/1, QK_SCALE, nullptr, Qb, Kb, Vtb);

  // attention: H * (L/64) blocks of 128 threads
  attn_kernel<<<H_DIM * (L_DIM / 64), 128, 0, stream>>>(Qb, Kb, Vtb, Ob);

  // out projection: M=4096, N=1024, K=1024 -> 32 x 8 tiles, f32 output
  gemm_bf16<<<32 * 8, 256, 0, stream>>>(Ob, wob, L_DIM, E_DIM, E_DIM, 8,
                                        /*mode=*/0, 1.0f, out, nullptr, nullptr, nullptr);
}
